// ViewTransformerLSS_14396730376668
// MI455X (gfx1250) — matrix operand order repaired and checked by reading, not compiled, not run on hardware
//
#include <hip/hip_runtime.h>

// ---------------------------------------------------------------------------
// LSS view transformer for MI455X (gfx1250, wave32, WMMA), round 3.
//
// Round-2 histogram showed clean conv codegen (s_clause'd global_load_b128
// feeding straight-line v_wmma_f32_16x16x32_f16). The only remaining
// exec-mask branching was the depth-head epilogue's row<105 guard; this
// round pads cd to a 112-row stride (7 full 16-row m-tiles) and pre-pads
// the bias vector, making every store unconditional.
// ---------------------------------------------------------------------------

typedef __attribute__((ext_vector_type(16))) _Float16 v16h;
typedef __attribute__((ext_vector_type(8)))  _Float16 v8h;
typedef __attribute__((ext_vector_type(8)))  float    v8f;

namespace {
constexpr int kB    = 8;
constexpr int kC    = 64;
constexpr int kH    = 32;
constexpr int kW    = 88;
constexpr int kHW   = kH * kW;        // 2816
constexpr int kD    = 41;
constexpr int kCD   = kC + kD;        // 105 real depth-head outputs
constexpr int kMT   = 7;              // m-tiles
constexpr int kCDP  = kMT * 16;       // 112 padded rows in cd scratch
constexpr int kBEV  = 256;
constexpr int kPad  = kBEV + 2;       // 258 (1-px zero halo)
constexpr float kXB0    = -51.2f;
constexpr float kInvRes = 1.0f / 0.4f;
constexpr float kBnEps  = 1e-5f;
}

// A-fragment (16x32 f16, MxK) K index for lane/element (contiguous: k = 16*(lane/16) + e).
__device__ __forceinline__ int a_k(int lane, int e) {
    int base = (lane < 16) ? 0 : 16;
    return base + e;
}

// ===========================================================================
// Pack kernels
// ===========================================================================

// w_depth (105x64 f32) -> wdp[mt][ks][lane][e] f16, rows >=105 zero-padded.
__global__ void pack_wd(const float* __restrict__ wd, _Float16* __restrict__ wdp)
{
    const int idx = blockIdx.x * blockDim.x + threadIdx.x;
    if (idx >= kMT * 2 * 32 * 16) return;
    const int e    = idx & 15;
    const int lane = (idx >> 4) & 31;
    const int ks   = (idx >> 9) & 1;
    const int mt   = idx >> 10;
    const int row  = mt * 16 + (lane & 15);
    const float val = (row < kCD) ? wd[row * kC + ks * 32 + a_k(lane, e)] : 0.0f;
    wdp[idx] = (_Float16)val;
}

// b_depth (105 f32) -> bdp (112 f32, zero-padded).
__global__ void pack_bd(const float* __restrict__ bd, float* __restrict__ bdp)
{
    const int i = threadIdx.x;
    if (i < kCDP) bdp[i] = (i < kCD) ? bd[i] : 0.0f;
}

// img (NCHW f32) -> imgp[b][n][c] f16 (pixel-major: B-fragment friendly).
__global__ void pack_img(const float* __restrict__ img, _Float16* __restrict__ imgp)
{
    const int idx = blockIdx.x * blockDim.x + threadIdx.x;
    if (idx >= kB * kHW * kC) return;
    const int c = idx & 63;
    const int n = (idx >> 6) % kHW;
    const int b = (idx >> 6) / kHW;
    imgp[idx] = (_Float16)img[((size_t)b * kC + c) * kHW + n];
}

// conv weights (64x64x3x3 f32) -> wpk[ot][ky][kx][ks][lane][e] f16.
__global__ void pack_conv_w(const float* __restrict__ w, _Float16* __restrict__ wpk)
{
    const int idx = blockIdx.x * blockDim.x + threadIdx.x;
    if (idx >= 4 * 3 * 3 * 2 * 32 * 16) return;
    const int e    = idx & 15;
    const int lane = (idx >> 4) & 31;
    const int ks   = (idx >> 9) & 1;
    int t = idx >> 10;
    const int kx = t % 3;  t /= 3;
    const int ky = t % 3;  t /= 3;
    const int ot = t;
    const int oc = ot * 16 + (lane & 15);
    const int ci = ks * 32 + a_k(lane, e);
    wpk[idx] = (_Float16)w[((size_t)(oc * kC + ci)) * 9 + ky * 3 + kx];
}

// BN -> per-channel (scale, shift).
__global__ void pack_bn(const float* __restrict__ g, const float* __restrict__ be,
                        const float* __restrict__ m, const float* __restrict__ v,
                        float* __restrict__ bnA, float* __restrict__ bnB)
{
    const int i = threadIdx.x;
    if (i < kC) {
        const float inv = g[i] * rsqrtf(v[i] + kBnEps);
        bnA[i] = inv;
        bnB[i] = be[i] - m[i] * inv;
    }
}

// bev (NCHW f32) -> p1 (padded NHWC f16). Halo pre-zeroed by memset.
__global__ void pack_bev(const float* __restrict__ bev, _Float16* __restrict__ p1)
{
    const int idx = blockIdx.x * blockDim.x + threadIdx.x;
    if (idx >= kB * kC * kBEV * kBEV) return;
    const int c = idx & 63;
    const int x = (idx >> 6) & 255;
    const int y = (idx >> 14) & 255;
    const int b = idx >> 22;
    p1[(((size_t)b * kPad + (y + 1)) * kPad + (x + 1)) * kC + c] =
        (_Float16)bev[(((size_t)b * kC + c) * kBEV + y) * kBEV + x];
}

// ===========================================================================
// Stage A: depth head GEMM. Wave per 16x16 tile, 2 WMMAs, fully branch-free.
// cd scratch has 112-row stride, so stores need no guards.
// ===========================================================================
__global__ void lss_depth_head(const _Float16* __restrict__ imgp, // [B][2816][64]
                               const _Float16* __restrict__ wdp,  // [7][2][32][16]
                               const float* __restrict__ bdp,     // [112]
                               float* __restrict__ cd)            // [B][112][2816]
{
    const int wid  = (blockIdx.x * blockDim.x + threadIdx.x) >> 5;
    const int lane = threadIdx.x & 31;
    if (wid >= (kHW / 16) * kMT * kB) return;
    const int nt = wid % (kHW / 16);
    const int t  = wid / (kHW / 16);
    const int mt = t % kMT;
    const int b  = t / kMT;
    const int n0 = nt * 16;
    const int nB = lane & 15;
    const int halfB = (lane < 16) ? 0 : 16;

    v8f acc = {};
#pragma unroll
    for (int ks = 0; ks < 2; ++ks) {
        const v16h a  = *(const v16h*)(wdp + (((mt * 2 + ks) * 32 + lane) << 4));
        const v16h bf = *(const v16h*)(imgp + (((size_t)b * kHW + n0 + nB) << 6)
                                            + ks * 32 + halfB);
        acc = __builtin_amdgcn_wmma_f32_16x16x32_f16(false, a, false, bf,
                                                     (short)0, acc, false, false);
    }
    const int row0 = mt * 16 + ((lane < 16) ? 0 : 8);
#pragma unroll
    for (int r = 0; r < 8; ++r) {
        cd[((size_t)b * kCDP + row0 + r) * kHW + (n0 + nB)] = acc[r] + bdp[row0 + r];
    }
}

// ===========================================================================
// Stage B: softmax over the 41 depth rows of cd, in place.
// ===========================================================================
__global__ void lss_depth_softmax(float* __restrict__ cd)
{
    const int idx = blockIdx.x * blockDim.x + threadIdx.x;
    if (idx >= kB * kHW) return;
    const int b = idx / kHW, n = idx % kHW;
    float* p = cd + ((size_t)b * kCDP + kC) * kHW + n;

    float mx = -3.4e38f;
    for (int d = 0; d < kD; ++d) mx = fmaxf(mx, p[(size_t)d * kHW]);
    float s = 0.0f;
    for (int d = 0; d < kD; ++d) s += __expf(p[(size_t)d * kHW] - mx);
    const float inv = 1.0f / s;
    for (int d = 0; d < kD; ++d)
        p[(size_t)d * kHW] = __expf(p[(size_t)d * kHW] - mx) * inv;
}

// ===========================================================================
// Stage C: geometry + splat. Wave per frustum point, lanes sweep channels.
// ===========================================================================
__global__ void lss_splat(const float* __restrict__ cd,
                          const float* __restrict__ intr,
                          const float* __restrict__ extr,
                          float* __restrict__ bev)
{
    const int wid  = (blockIdx.x * blockDim.x + threadIdx.x) >> 5;
    const int lane = threadIdx.x & 31;
    if (wid >= kB * kD * kHW) return;

    const int b   = wid / (kD * kHW);
    const int rem = wid % (kD * kHW);
    const int d   = rem / kHW;
    const int n   = rem % kHW;
    const float px = (float)(n % kW);
    const float py = (float)(n / kW);

    const float* K = intr + b * 9;
    const float k00=K[0],k01=K[1],k02=K[2],k10=K[3],k11=K[4],k12=K[5],k20=K[6],k21=K[7],k22=K[8];
    const float det = k00*(k11*k22 - k12*k21) - k01*(k10*k22 - k12*k20) + k02*(k10*k21 - k11*k20);
    const float id  = 1.0f / det;
    const float i00=(k11*k22-k12*k21)*id, i01=(k02*k21-k01*k22)*id, i02=(k01*k12-k02*k11)*id;
    const float i10=(k12*k20-k10*k22)*id, i11=(k00*k22-k02*k20)*id, i12=(k02*k10-k00*k12)*id;
    const float i20=(k10*k21-k11*k20)*id, i21=(k01*k20-k00*k21)*id, i22=(k00*k11-k01*k10)*id;

    const float depth = 1.0f + 1.1f * (float)d;
    const float cx = depth * (i00*px + i01*py + i02);
    const float cy = depth * (i10*px + i11*py + i12);
    const float cz = depth * (i20*px + i21*py + i22);

    const float* E = extr + b * 16;
    const float gx = E[0]*cx + E[1]*cy + E[2]*cz + E[3];
    const float gy = E[4]*cx + E[5]*cy + E[6]*cz + E[7];

    const int xi = (int)((gx - kXB0) * kInvRes);
    const int yi = (int)((gy - kXB0) * kInvRes);
    if (xi < 0 || xi >= kBEV || yi < 0 || yi >= kBEV) return;

    const float w = cd[((size_t)b * kCDP + kC + d) * kHW + n];
    const size_t base = ((size_t)b * kC) * kBEV * kBEV + (size_t)yi * kBEV + xi;
#pragma unroll
    for (int it = 0; it < 2; ++it) {
        const int c = lane + it * 32;
        const float f = w * cd[((size_t)b * kCDP + c) * kHW + n];
        atomicAdd(bev + base + (size_t)c * kBEV * kBEV, f);
    }
}

// ===========================================================================
// Stage D/E: 3x3 conv + BN + ReLU, implicit GEMM. Wave per 16oc x 16px tile.
// 9 taps x 2 K-steps = 18 straight-line WMMAs, each fed by two 32B loads.
// ===========================================================================
template <bool F16OUT>
__global__ void lss_conv_wmma(const _Float16* __restrict__ xin, // [B][258][258][64] f16
                              const _Float16* __restrict__ wpk, // [4][3][3][2][32][16]
                              const float* __restrict__ bnA,
                              const float* __restrict__ bnB,
                              _Float16* __restrict__ yf16,      // padded NHWC (F16OUT)
                              float* __restrict__ yf32)         // NCHW (!F16OUT)
{
    const int wid  = (blockIdx.x * blockDim.x + threadIdx.x) >> 5;
    const int lane = threadIdx.x & 31;
    // wid bits: [ct:4][row:8][ot:2][b:3] -> 2^17 tiles
    const int ct  = wid & 15;
    const int row = (wid >> 4) & 255;
    const int ot  = (wid >> 12) & 3;
    const int b   = wid >> 14;
    const int col0  = ct * 16;
    const int nB    = lane & 15;
    const int halfB = (lane < 16) ? 0 : 16;

    v8f acc = {};
#pragma unroll
    for (int ky = 0; ky < 3; ++ky) {
#pragma unroll
        for (int kx = 0; kx < 3; ++kx) {
            const size_t pix = ((size_t)(b * kPad + row + ky) * kPad
                                + (col0 + nB + kx)) << 6;   // *64 channels
#pragma unroll
            for (int ks = 0; ks < 2; ++ks) {
                const v16h a  = *(const v16h*)(wpk +
                    ((((((ot * 3 + ky) * 3 + kx) * 2 + ks) * 32) + lane) << 4));
                const v16h bf = *(const v16h*)(xin + pix + ks * 32 + halfB);
                acc = __builtin_amdgcn_wmma_f32_16x16x32_f16(false, a, false, bf,
                                                             (short)0, acc, false, false);
            }
        }
    }

    const int half8 = (lane < 16) ? 0 : 8;
    if (F16OUT) {
        v8h outv;
#pragma unroll
        for (int r = 0; r < 8; ++r) {
            const int oc = ot * 16 + half8 + r;
            outv[r] = (_Float16)fmaxf(acc[r] * bnA[oc] + bnB[oc], 0.0f);
        }
        _Float16* dst = yf16 + ((((size_t)(b * kPad + row + 1) * kPad)
                                 + (col0 + 1 + nB)) << 6) + ot * 16 + half8;
        *(v8h*)dst = outv;
    } else {
#pragma unroll
        for (int r = 0; r < 8; ++r) {
            const int oc = ot * 16 + half8 + r;
            const float val = fmaxf(acc[r] * bnA[oc] + bnB[oc], 0.0f);
            yf32[(((size_t)b * kC + oc) * kBEV + row) * kBEV + (col0 + nB)] = val;
        }
    }
}

// ===========================================================================
// Host launcher
// ===========================================================================
extern "C" void kernel_launch(void* const* d_in, const int* in_sizes, int n_in,
                              void* d_out, int out_size, void* d_ws, size_t ws_size,
                              hipStream_t stream)
{
    (void)in_sizes; (void)n_in; (void)out_size; (void)ws_size;

    const float* img  = (const float*)d_in[0];
    const float* intr = (const float*)d_in[1];
    const float* extr = (const float*)d_in[2];
    const float* wd   = (const float*)d_in[3];
    const float* bd   = (const float*)d_in[4];
    const float* w1   = (const float*)d_in[5];
    const float* g1   = (const float*)d_in[6];
    const float* be1  = (const float*)d_in[7];
    const float* m1   = (const float*)d_in[8];
    const float* v1   = (const float*)d_in[9];
    const float* w2   = (const float*)d_in[10];
    const float* g2   = (const float*)d_in[11];
    const float* be2  = (const float*)d_in[12];
    const float* m2   = (const float*)d_in[13];
    const float* v2   = (const float*)d_in[14];
    float* out = (float*)d_out;

    // ---- workspace carve (all regions 256B aligned) ----
    char* wsp = (char*)d_ws;
    auto carve = [&wsp](size_t bytes) {
        char* p = wsp;
        wsp += (bytes + 255) & ~(size_t)255;
        return p;
    };
    const size_t cdElems  = (size_t)kB * kCDP * kHW;
    const size_t bevElems = (size_t)kB * kC * kBEV * kBEV;
    const size_t padElems = (size_t)kB * kPad * kPad * kC;

    float*    cd   = (float*)   carve(cdElems  * 4);
    float*    bev  = (float*)   carve(bevElems * 4);
    _Float16* p1   = (_Float16*)carve(padElems * 2);
    _Float16* p2   = (_Float16*)carve(padElems * 2);
    _Float16* imgp = (_Float16*)carve((size_t)kB * kHW * kC * 2);
    _Float16* wdp  = (_Float16*)carve((size_t)kMT * 2 * 32 * 16 * 2);
    _Float16* wpk1 = (_Float16*)carve((size_t)4 * 9 * 2 * 32 * 16 * 2);
    _Float16* wpk2 = (_Float16*)carve((size_t)4 * 9 * 2 * 32 * 16 * 2);
    float*    bdp  = (float*)   carve(kCDP * 4);
    float*    bn1A = (float*)   carve(kC * 4);
    float*    bn1B = (float*)   carve(kC * 4);
    float*    bn2A = (float*)   carve(kC * 4);
    float*    bn2B = (float*)   carve(kC * 4);

    // ---- packing (independent, cheap) ----
    {
        const int n = kMT * 2 * 32 * 16;
        pack_wd<<<(n + 255) / 256, 256, 0, stream>>>(wd, wdp);
    }
    pack_bd<<<1, kCDP, 0, stream>>>(bd, bdp);
    {
        const int n = kB * kHW * kC;
        pack_img<<<(n + 255) / 256, 256, 0, stream>>>(img, imgp);
    }
    {
        const int n = 4 * 9 * 2 * 32 * 16;
        pack_conv_w<<<(n + 255) / 256, 256, 0, stream>>>(w1, wpk1);
        pack_conv_w<<<(n + 255) / 256, 256, 0, stream>>>(w2, wpk2);
    }
    pack_bn<<<1, 64, 0, stream>>>(g1, be1, m1, v1, bn1A, bn1B);
    pack_bn<<<1, 64, 0, stream>>>(g2, be2, m2, v2, bn2A, bn2B);

    // zero BEV accumulator and both padded activation buffers (halo must be 0)
    hipMemsetAsync(bev, 0, bevElems * 4, stream);
    hipMemsetAsync(p1, 0, padElems * 2, stream);
    hipMemsetAsync(p2, 0, padElems * 2, stream);

    // ---- stage A: depth head (WMMA) ----
    {
        const long long waves = (long long)(kHW / 16) * kMT * kB;  // 9856
        lss_depth_head<<<(unsigned)((waves * 32 + 255) / 256), 256, 0, stream>>>(
            imgp, wdp, bdp, cd);
    }
    // ---- stage B: softmax ----
    {
        const int threads = kB * kHW;
        lss_depth_softmax<<<(threads + 255) / 256, 256, 0, stream>>>(cd);
    }
    // ---- stage C: splat ----
    {
        const long long waves = (long long)kB * kD * kHW;          // 923,648
        lss_splat<<<(unsigned)((waves * 32 + 255) / 256), 256, 0, stream>>>(
            cd, intr, extr, bev);
    }
    // ---- repack bev into padded NHWC f16 ----
    {
        const long long n = (long long)kB * kC * kBEV * kBEV;
        pack_bev<<<(unsigned)((n + 255) / 256), 256, 0, stream>>>(bev, p1);
    }
    // ---- stage D: conv1 (WMMA) -> p2 (f16 NHWC padded) ----
    {
        const long long waves = (long long)kB * 4 * kBEV * (kBEV / 16); // 131072
        lss_conv_wmma<true><<<(unsigned)((waves * 32 + 255) / 256), 256, 0, stream>>>(
            p1, wpk1, bn1A, bn1B, p2, nullptr);
    }
    // ---- stage E: conv2 (WMMA) -> d_out (f32 NCHW) ----
    {
        const long long waves = (long long)kB * 4 * kBEV * (kBEV / 16);
        lss_conv_wmma<false><<<(unsigned)((waves * 32 + 255) / 256), 256, 0, stream>>>(
            p2, wpk2, bn2A, bn2B, nullptr, out);
    }
}
